// DorefaDense_38268158607678
// MI455X (gfx1250) — 8-bit matrix operand order repaired and checked by reading, compiled, not run on hardware
//
#include <hip/hip_runtime.h>

typedef __attribute__((ext_vector_type(16))) int   v16i;
typedef __attribute__((ext_vector_type(8)))  float v8f;

#define BATCH   8192
#define IN_CH   4096
#define NUNITS  4096

static const size_t AQ_BYTES = (size_t)BATCH * IN_CH;
static const size_t BT_BYTES = (size_t)IN_CH * NUNITS;

#define FP8_INT_LUT 0x4E4C4A4844403800ull
#define FP8_POS_ONE 0x38
#define FP8_NEG_ONE 0xB8

__global__ __launch_bounds__(32) void init_scale_k(float* e) {
    if (threadIdx.x == 0) e[0] = 0.0f;
}

__global__ __launch_bounds__(256) void absmean_reduce_k(const float* __restrict__ W,
                                                        float* __restrict__ e) {
    __shared__ float sm[256];
    const size_t n = (size_t)IN_CH * NUNITS;
    const size_t stride = (size_t)gridDim.x * blockDim.x;
    float s = 0.0f;
    for (size_t i = (size_t)blockIdx.x * blockDim.x + threadIdx.x; i < n; i += stride)
        s += fabsf(W[i]);
    sm[threadIdx.x] = s;
    __syncthreads();
    for (int off = 128; off > 0; off >>= 1) {
        if ((int)threadIdx.x < off) sm[threadIdx.x] += sm[threadIdx.x + off];
        __syncthreads();
    }
    if (threadIdx.x == 0) atomicAdd(e, sm[0]);
}

__global__ __launch_bounds__(256) void quant_act_k(const float4* __restrict__ x,
                                                   uchar4* __restrict__ q) {
    size_t i = (size_t)blockIdx.x * blockDim.x + threadIdx.x;
    float4 v = x[i];
    int kx = __float2int_rn(fminf(1.0f, fabsf(v.x)) * 7.0f);
    int ky = __float2int_rn(fminf(1.0f, fabsf(v.y)) * 7.0f);
    int kz = __float2int_rn(fminf(1.0f, fabsf(v.z)) * 7.0f);
    int kw = __float2int_rn(fminf(1.0f, fabsf(v.w)) * 7.0f);
    uchar4 o;
    o.x = (unsigned char)(FP8_INT_LUT >> (kx * 8));
    o.y = (unsigned char)(FP8_INT_LUT >> (ky * 8));
    o.z = (unsigned char)(FP8_INT_LUT >> (kz * 8));
    o.w = (unsigned char)(FP8_INT_LUT >> (kw * 8));
    q[i] = o;
}

__global__ void quant_wt_k(const float* __restrict__ W,
                           unsigned char* __restrict__ Bt) {
    __shared__ unsigned char tile[32][33];
    const int n0 = blockIdx.x * 32;
    const int k0 = blockIdx.y * 32;
    for (int r = threadIdx.y; r < 32; r += 8) {
        float w = W[(size_t)(k0 + r) * NUNITS + n0 + threadIdx.x];
        tile[r][threadIdx.x] = (w > 0.0f) ? (unsigned char)FP8_POS_ONE
                              : (w < 0.0f) ? (unsigned char)FP8_NEG_ONE
                                           : (unsigned char)0x00;
    }
    __syncthreads();
    for (int r = threadIdx.y; r < 32; r += 8) {
        Bt[(size_t)(n0 + r) * IN_CH + k0 + threadIdx.x] = tile[threadIdx.x][r];
    }
}

union AFrag { v16i v; unsigned long long q[8]; };
union BFrag { v16i v; int4 x[4]; };

__global__ __launch_bounds__(256) void gemm_dorefa_k(
        const unsigned char* __restrict__ Aq,
        const unsigned char* __restrict__ Bt,
        const float* __restrict__ bias,
        const float* __restrict__ esum,
        float* __restrict__ out) {
    const int K = IN_CH;
    const int wave  = threadIdx.x >> 5;
    const int lane  = threadIdx.x & 31;
    const int tile  = blockIdx.x * 8 + wave;
    const int tilesN = NUNITS / 64;
    const int row0 = (tile / tilesN) * 64;
    const int col0 = (tile % tilesN) * 64;
    const int lhalf = lane >> 4;
    const int l16   = lane & 15;

    v8f acc[4][4];
    const v8f vzero = {0.f, 0.f, 0.f, 0.f, 0.f, 0.f, 0.f, 0.f};
#pragma unroll
    for (int mi = 0; mi < 4; ++mi)
#pragma unroll
        for (int ni = 0; ni < 4; ++ni)
            acc[mi][ni] = vzero;

    const unsigned char* arow[4];
    const unsigned char* brow[4];
#pragma unroll
    for (int mi = 0; mi < 4; ++mi)
        arow[mi] = Aq + (size_t)(row0 + mi * 16 + l16) * K + lhalf * 16;
#pragma unroll
    for (int ni = 0; ni < 4; ++ni)
        brow[ni] = Bt + (size_t)(col0 + ni * 16 + l16) * K + lhalf * 16;

    for (int k0 = 0; k0 < K; k0 += 128) {
        BFrag bf[4];
#pragma unroll
        for (int ni = 0; ni < 4; ++ni) {
            const unsigned char* bp = brow[ni] + k0;
            bf[ni].x[0] = *(const int4*)(bp);
            bf[ni].x[1] = *(const int4*)(bp + 32);
            bf[ni].x[2] = *(const int4*)(bp + 64);
            bf[ni].x[3] = *(const int4*)(bp + 96);
        }
#pragma unroll
        for (int mi = 0; mi < 4; ++mi) {
            AFrag af;
            const unsigned char* ap = arow[mi] + k0;
#pragma unroll
            for (int j = 0; j < 8; ++j)
                af.q[j] = *(const unsigned long long*)(ap + 32 * (j >> 1) + 8 * (j & 1));
#pragma unroll
            for (int ni = 0; ni < 4; ++ni) {
                acc[mi][ni] = __builtin_amdgcn_wmma_f32_16x16x128_fp8_fp8(
                    af.v, bf[ni].v, (short)0, acc[mi][ni],
                     false,  false);
            }
        }
    }

    const float scale = esum[0] * (1.0f / ((float)IN_CH * (float)NUNITS * 7.0f));
#pragma unroll
    for (int mi = 0; mi < 4; ++mi) {
#pragma unroll
        for (int ni = 0; ni < 4; ++ni) {
            const int col = col0 + ni * 16 + l16;
            const float bv = bias[col];
#pragma unroll
            for (int r = 0; r < 8; ++r) {
                const int row = row0 + mi * 16 + lhalf * 8 + r;
                __builtin_nontemporal_store(scale * acc[mi][ni][r] + bv,
                                            &out[(size_t)row * NUNITS + col]);
            }
        }
    }
}

extern "C" void kernel_launch(void* const* d_in, const int* in_sizes, int n_in,
                              void* d_out, int out_size, void* d_ws, size_t ws_size,
                              hipStream_t stream) {
    (void)in_sizes; (void)n_in; (void)out_size; (void)ws_size;
    const float* x    = (const float*)d_in[0];
    const float* W    = (const float*)d_in[1];
    const float* bias = (const float*)d_in[2];
    float* out = (float*)d_out;

    unsigned char* Aq = (unsigned char*)d_ws;
    unsigned char* Bt = (unsigned char*)d_ws + AQ_BYTES;
    float*       esum = (float*)((char*)d_ws + AQ_BYTES + BT_BYTES);

    init_scale_k<<<1, 32, 0, stream>>>(esum);
    absmean_reduce_k<<<1024, 256, 0, stream>>>(W, esum);

    const int actVec = (BATCH * IN_CH) / 4;
    quant_act_k<<<actVec / 256, 256, 0, stream>>>((const float4*)x, (uchar4*)Aq);

    dim3 wtGrid(NUNITS / 32, IN_CH / 32);
    dim3 wtBlock(32, 8);
    quant_wt_k<<<wtGrid, wtBlock, 0, stream>>>(W, Bt);

    const int tiles = (BATCH / 64) * (NUNITS / 64);
    gemm_dorefa_k<<<tiles / 8, 256, 0, stream>>>(Aq, Bt, bias, esum, out);
}
